// MiMoAudioAttention_21964462752159
// MI455X (gfx1250) — matrix operand order repaired and checked by reading, not compiled, not run on hardware
//
#include <hip/hip_runtime.h>
#include <math.h>

// ---------------- problem constants ----------------
constexpr int Bb   = 2;
constexpr int Ss   = 2048;
constexpr int Hh   = 2048;
constexpr int NHq  = 16;
constexpr int NKVq = 4;
constexpr int HDq  = 128;
constexpr int MTOK = Bb * Ss;                       // 4096 tokens
constexpr int NQKV = NHq * HDq + 2 * NKVq * HDq;    // 3072
constexpr float THETA_F = 10000.0f;
constexpr float SCALE_F = 0.08838834764831845f;     // 128^-0.5

// ---------------- WMMA types ----------------
typedef __attribute__((ext_vector_type(16))) __bf16 v16bf;
typedef __attribute__((ext_vector_type(8)))  float  v8f;

// A-fragment (16x32 MxK, bf16): lane holds row m=lane%16; 16 contiguous
// K values starting at K = 16*half (half = lane/16).
__device__ __forceinline__ v16bf load_a_frag(const __bf16* row, int k, int hl) {
  union { v16bf v; uint4 u[2]; } r;
  r.u[0] = *(const uint4*)(row + k + 16 * hl);
  r.u[1] = *(const uint4*)(row + k + 16 * hl + 8);
  return r.v;
}
// B-fragment (32x16 KxN, bf16): lane holds col n=lane%16; 16 contiguous K
// values starting at K = 16*half. Caller passes a 32B-aligned pointer.
__device__ __forceinline__ v16bf load_b_frag(const __bf16* p) {
  union { v16bf v; uint4 u[2]; } r;
  r.u[0] = *(const uint4*)(p);
  r.u[1] = *(const uint4*)(p + 8);
  return r.v;
}

// ---------------- conversion / packing kernels ----------------
__global__ __launch_bounds__(256) void cvt_f32_bf16(const float* __restrict__ src,
                                                    __bf16* __restrict__ dst, int n) {
  for (int i = blockIdx.x * 256 + threadIdx.x; i < n; i += gridDim.x * 256)
    dst[i] = (__bf16)src[i];
}

// dst[n][K] = (bf16) src[k][N]  (transpose + convert); dst row length = K
__global__ __launch_bounds__(256) void cvt_wt_t(const float* __restrict__ src,
                                                __bf16* __restrict__ dst,
                                                int K, int N) {
  int total = K * N;
  for (int i = blockIdx.x * 256 + threadIdx.x; i < total; i += gridDim.x * 256) {
    int n = i / K, k = i - n * K;
    dst[i] = (__bf16)src[(size_t)k * N + n];
  }
}

__global__ __launch_bounds__(256) void concat_bias(const float* __restrict__ bq,
                                                   const float* __restrict__ bk,
                                                   const float* __restrict__ bv,
                                                   float* __restrict__ out) {
  int i = blockIdx.x * 256 + threadIdx.x;
  if (i < 2048)      out[i] = bq[i];
  else if (i < 2560) out[i] = bk[i - 2048];
  else if (i < 3072) out[i] = bv[i - 2560];
}

// ---------------- bf16 WMMA GEMM: C[M][N] f32 = A[M][K] @ Bt[N][K] + bias ---
// block = 256 thr = 8 waves (2 M x 4 N), wave tile = 32x64, block tile 64x256
__global__ __launch_bounds__(256) void gemm_bf16(const __bf16* __restrict__ A,
                                                 const __bf16* __restrict__ Bt,
                                                 const float* __restrict__ bias,
                                                 float* __restrict__ C,
                                                 int Ndim, int K) {
  const int lane = threadIdx.x & 31;
  const int wv   = threadIdx.x >> 5;
  const int hl   = lane >> 4;
  const int l16  = lane & 15;
  const int mBase = blockIdx.y * 64 + (wv & 1) * 32;
  const int nBase = blockIdx.x * 256 + (wv >> 1) * 64;

  v8f acc[2][4] = {};

  const __bf16* Arow0 = A + (size_t)(mBase + l16) * K;
  const __bf16* Arow1 = Arow0 + (size_t)16 * K;
  const __bf16* Brow[4];
#pragma unroll
  for (int j = 0; j < 4; ++j)
    Brow[j] = Bt + (size_t)(nBase + j * 16 + l16) * K + 16 * hl;

  for (int k = 0; k < K; k += 32) {
    v16bf a0 = load_a_frag(Arow0, k, hl);
    v16bf a1 = load_a_frag(Arow1, k, hl);
    v16bf b[4];
#pragma unroll
    for (int j = 0; j < 4; ++j) b[j] = load_b_frag(Brow[j] + k);
#pragma unroll
    for (int j = 0; j < 4; ++j) {
      acc[0][j] = __builtin_amdgcn_wmma_f32_16x16x32_bf16(false, a0, false, b[j],
                                                          (short)0, acc[0][j], false, false);
      acc[1][j] = __builtin_amdgcn_wmma_f32_16x16x32_bf16(false, a1, false, b[j],
                                                          (short)0, acc[1][j], false, false);
    }
  }

#pragma unroll
  for (int i = 0; i < 2; ++i)
#pragma unroll
    for (int j = 0; j < 4; ++j) {
      int col = nBase + j * 16 + l16;
      float bvv = bias ? bias[col] : 0.0f;
#pragma unroll
      for (int r = 0; r < 8; ++r) {
        size_t row = (size_t)(mBase + i * 16 + 8 * hl + r);
        C[row * Ndim + col] = acc[i][j][r] + bvv;
      }
    }
}

// ---------------- RoPE + pack to bf16 ----------------
// qkv f32 [4096][3072] -> Q[b][h][s][128], K[b][kv][s][128], Vt[b][kv][128][s]
__global__ __launch_bounds__(256) void rope_pack(const float* __restrict__ qkv,
                                                 const int* __restrict__ positions,
                                                 __bf16* __restrict__ Q,
                                                 __bf16* __restrict__ Kc,
                                                 __bf16* __restrict__ Vt) {
  int t = blockIdx.x;                 // token 0..4095
  int b = t / Ss, s = t - b * Ss;
  float pos = (float)positions[s];
  const float* row = qkv + (size_t)t * NQKV;
  int tid = threadIdx.x;

  for (int it = tid; it < NHq * 64; it += 256) {          // Q rotate
    int h = it >> 6, d = it & 63;
    float fr = pos * __powf(THETA_F, -2.0f * d / HDq);
    float c = __cosf(fr), sn = __sinf(fr);
    float x1 = row[h * HDq + d], x2 = row[h * HDq + 64 + d];
    size_t o = (((size_t)b * NHq + h) * Ss + s) * HDq;
    Q[o + d]      = (__bf16)(x1 * c - x2 * sn);
    Q[o + 64 + d] = (__bf16)(x2 * c + x1 * sn);
  }
  for (int it = tid; it < NKVq * 64; it += 256) {         // K rotate
    int h = it >> 6, d = it & 63;
    float fr = pos * __powf(THETA_F, -2.0f * d / HDq);
    float c = __cosf(fr), sn = __sinf(fr);
    float x1 = row[2048 + h * HDq + d], x2 = row[2048 + h * HDq + 64 + d];
    size_t o = (((size_t)b * NKVq + h) * Ss + s) * HDq;
    Kc[o + d]      = (__bf16)(x1 * c - x2 * sn);
    Kc[o + 64 + d] = (__bf16)(x2 * c + x1 * sn);
  }
  for (int it = tid; it < NKVq * HDq; it += 256) {        // V transpose
    int h = it >> 7, d = it & 127;
    Vt[(((size_t)b * NKVq + h) * HDq + d) * Ss + s] = (__bf16)row[2560 + h * HDq + d];
  }
}

// ---------------- flash attention (wave32, WMMA bf16) ----------------
// one wave = one 16-query block; key tiles of 32; online softmax.
__global__ __launch_bounds__(256) void flash_attn(const __bf16* __restrict__ Q,
                                                  const __bf16* __restrict__ Kc,
                                                  const __bf16* __restrict__ Vt,
                                                  __bf16* __restrict__ Oout) {
  __shared__ __align__(16) __bf16 pshm[8][16][32];   // per-wave P bounce
  const int lane = threadIdx.x & 31;
  const int wv   = threadIdx.x >> 5;
  const int hl   = lane >> 4;
  const int l16  = lane & 15;

  const int bh = blockIdx.y;                          // 0..31
  const int b  = bh / NHq, h = bh - b * NHq;
  const int kv = h / (NHq / NKVq);
  const int q0 = (blockIdx.x * 8 + wv) * 16;

  const __bf16* Qbase = Q  + ((size_t)b * NHq  + h ) * Ss * HDq;
  const __bf16* Kbase = Kc + ((size_t)b * NKVq + kv) * Ss * HDq;
  const __bf16* Vbase = Vt + ((size_t)b * NKVq + kv) * HDq * (size_t)Ss;

  // preload Q fragments (4 K-steps of 32 over HD=128)
  v16bf qf[4];
  const __bf16* qrow = Qbase + (size_t)(q0 + l16) * HDq;
#pragma unroll
  for (int ks = 0; ks < 4; ++ks) qf[ks] = load_a_frag(qrow, ks * 32, hl);

  v8f o[8] = {};
  float mrow[8], lrow[8];
#pragma unroll
  for (int r = 0; r < 8; ++r) { mrow[r] = -1e30f; lrow[r] = 0.0f; }

  const int kmax = q0 + 15;
  for (int k0 = 0; k0 <= kmax; k0 += 32) {
    // ---- S = Q K^T (16x32) ----
    v8f sacc[2] = {};
#pragma unroll
    for (int nt = 0; nt < 2; ++nt) {
      const __bf16* krow = Kbase + (size_t)(k0 + nt * 16 + l16) * HDq + 16 * hl;
#pragma unroll
      for (int ks = 0; ks < 4; ++ks) {
        v16bf kf = load_b_frag(krow + ks * 32);
        sacc[nt] = __builtin_amdgcn_wmma_f32_16x16x32_bf16(false, qf[ks], false, kf,
                                                           (short)0, sacc[nt], false, false);
      }
    }
    // ---- scale + causal mask + online softmax ----
    float alpha[8];
#pragma unroll
    for (int r = 0; r < 8; ++r) {
      int qpos = q0 + 8 * hl + r;
      float rmax = -1e30f;
#pragma unroll
      for (int nt = 0; nt < 2; ++nt) {
        int kpos = k0 + nt * 16 + l16;
        float sv = sacc[nt][r] * SCALE_F;
        sv = (kpos <= qpos) ? sv : -1e30f;
        sacc[nt][r] = sv;
        rmax = fmaxf(rmax, sv);
      }
#pragma unroll
      for (int off = 8; off; off >>= 1) rmax = fmaxf(rmax, __shfl_xor(rmax, off, 32));
      float mnew = fmaxf(mrow[r], rmax);
      alpha[r] = __expf(mrow[r] - mnew);
      mrow[r] = mnew;
      float rsum = 0.0f;
#pragma unroll
      for (int nt = 0; nt < 2; ++nt) {
        float p = __expf(sacc[nt][r] - mnew);
        sacc[nt][r] = p;
        rsum += p;
      }
#pragma unroll
      for (int off = 8; off; off >>= 1) rsum += __shfl_xor(rsum, off, 32);
      lrow[r] = lrow[r] * alpha[r] + rsum;
    }
    // rescale O accumulators
#pragma unroll
    for (int nt = 0; nt < 8; ++nt)
#pragma unroll
      for (int r = 0; r < 8; ++r) o[nt][r] *= alpha[r];

    // ---- P: C-layout -> LDS -> A-layout ----
#pragma unroll
    for (int nt = 0; nt < 2; ++nt)
#pragma unroll
      for (int r = 0; r < 8; ++r)
        pshm[wv][8 * hl + r][nt * 16 + l16] = (__bf16)sacc[nt][r];
    // intra-wave LDS ordering: compiler inserts dscnt waits
    union { v16bf v; uint4 u[2]; } pr;
    pr.u[0] = *(const uint4*)&pshm[wv][l16][16 * hl];
    pr.u[1] = *(const uint4*)&pshm[wv][l16][16 * hl + 8];
    v16bf pf = pr.v;

    // ---- O += P @ V ----
#pragma unroll
    for (int nt = 0; nt < 8; ++nt) {
      v16bf vf = load_b_frag(Vbase + (size_t)(nt * 16 + l16) * Ss + k0 + 16 * hl);
      o[nt] = __builtin_amdgcn_wmma_f32_16x16x32_bf16(false, pf, false, vf,
                                                      (short)0, o[nt], false, false);
    }
  }

  // ---- finalize: divide by l, store bf16 [token][NH*HD] ----
  float linv[8];
#pragma unroll
  for (int r = 0; r < 8; ++r) linv[r] = 1.0f / lrow[r];
#pragma unroll
  for (int nt = 0; nt < 8; ++nt)
#pragma unroll
    for (int r = 0; r < 8; ++r) {
      size_t row = (size_t)b * Ss + q0 + 8 * hl + r;
      Oout[row * (NHq * HDq) + h * HDq + nt * 16 + l16] = (__bf16)(o[nt][r] * linv[r]);
    }
}

// ---------------- host launcher ----------------
extern "C" void kernel_launch(void* const* d_in, const int* in_sizes, int n_in,
                              void* d_out, int out_size, void* d_ws, size_t ws_size,
                              hipStream_t stream) {
  (void)in_sizes; (void)n_in; (void)out_size; (void)ws_size;
  const float* hidden    = (const float*)d_in[0];
  const int*   positions = (const int*)d_in[1];
  const float* Wq = (const float*)d_in[2];
  const float* bq = (const float*)d_in[3];
  const float* Wk = (const float*)d_in[4];
  const float* bk = (const float*)d_in[5];
  const float* Wv = (const float*)d_in[6];
  const float* bv = (const float*)d_in[7];
  const float* Wo = (const float*)d_in[8];
  float* out = (float*)d_out;

  char* ws = (char*)d_ws;
  size_t off = 0;
  auto take = [&](size_t bytes) -> char* {
    char* p = ws + off;
    off = (off + bytes + 255) & ~(size_t)255;
    return p;
  };
  __bf16* Xb     = (__bf16*)take((size_t)MTOK * Hh * 2);          // hidden bf16
  __bf16* Wqkvt  = (__bf16*)take((size_t)NQKV * Hh * 2);          // [3072][2048]
  float*  biasq  = (float*) take((size_t)NQKV * 4);
  __bf16* Wot    = (__bf16*)take((size_t)Hh * (NHq * HDq) * 2);   // [2048][2048]
  float*  qkvf   = (float*) take((size_t)MTOK * NQKV * 4);        // [4096][3072]
  __bf16* Qb     = (__bf16*)take((size_t)Bb * NHq  * Ss * HDq * 2);
  __bf16* Kb     = (__bf16*)take((size_t)Bb * NKVq * Ss * HDq * 2);
  __bf16* Vtb    = (__bf16*)take((size_t)Bb * NKVq * HDq * Ss * 2);
  __bf16* attnb  = (__bf16*)take((size_t)MTOK * NHq * HDq * 2);

  // 1) conversions / transposes
  cvt_f32_bf16<<<4096, 256, 0, stream>>>(hidden, Xb, MTOK * Hh);
  cvt_wt_t<<<4096, 256, 0, stream>>>(Wq, Wqkvt,                         Hh, NHq  * HDq);
  cvt_wt_t<<<2048, 256, 0, stream>>>(Wk, Wqkvt + (size_t)2048 * Hh,     Hh, NKVq * HDq);
  cvt_wt_t<<<2048, 256, 0, stream>>>(Wv, Wqkvt + (size_t)2560 * Hh,     Hh, NKVq * HDq);
  cvt_wt_t<<<4096, 256, 0, stream>>>(Wo, Wot,                           NHq * HDq, Hh);
  concat_bias<<<12, 256, 0, stream>>>(bq, bk, bv, biasq);

  // 2) fused QKV projection: [4096][2048] @ [3072][2048]^T + bias
  gemm_bf16<<<dim3(NQKV / 256, MTOK / 64), 256, 0, stream>>>(Xb, Wqkvt, biasq, qkvf,
                                                             NQKV, Hh);
  // 3) RoPE + pack
  rope_pack<<<MTOK, 256, 0, stream>>>(qkvf, positions, Qb, Kb, Vtb);

  // 4) causal GQA flash attention
  flash_attn<<<dim3(Ss / 16 / 8, Bb * NHq), 256, 0, stream>>>(Qb, Kb, Vtb, attnb);

  // 5) output projection: [4096][2048] @ [2048][2048]^T -> d_out f32
  gemm_bf16<<<dim3(Hh / 256, MTOK / 64), 256, 0, stream>>>(attnb, Wot, nullptr, out,
                                                           Hh, Hh);
}
